// MultiHeadCSGA_20753281974717
// MI455X (gfx1250) — matrix operand order repaired and checked by reading, not compiled, not run on hardware
//
#include <hip/hip_runtime.h>
#include <cfloat>

// ---------------------------------------------------------------------------
// MultiHeadCSGA for MI455X (gfx1250, wave32, WMMA f16 16x16x32, f32 accum)
//   1. pool:      P_fg/P_bg[j][c] = sum_l w * X[c, p(s,l)]    (f16, 2560x256)
//   2. swizzle:   Wq/Ws -> exact WMMA B-frag layout (1 vload per frag)
//   3. qtrans:    X_query (c-major f32) -> qx16[p][c] f16 (LDS transpose)
//   4. GEMMs:     qf = qx @ Wq^T + bq ; c_s = (P @ Ws^T)*inv + bs*bfac (WMMA)
//   5. attention: 8-wave WG, context chunks async-staged to LDS (double buf),
//                 WMMA logits + online softmax + fg-sum, shfl merge
//   6. conv/GN/relu x3 tail -> d_out
// ---------------------------------------------------------------------------

typedef __attribute__((ext_vector_type(16))) _Float16 v16h;
typedef __attribute__((ext_vector_type(8)))  _Float16 v8h;
typedef __attribute__((ext_vector_type(8)))  float    v8f;
typedef __attribute__((ext_vector_type(4)))  int      v4i;

#define KIMG   5
#define CDIM   256
#define HW     4096
#define SPAT   256      // S = P*P
#define NJ     2560     // K*2*S
#define HEADS  8
#define DDIM   32
#define CHUNK  64       // attention j's staged per LDS chunk
#define NCHUNK (NJ / CHUNK)

// -------- gfx1250 async global->LDS path (probe via __has_builtin) ---------
#if defined(__has_builtin)
#if __has_builtin(__builtin_amdgcn_global_load_async_to_lds_b128) && \
    __has_builtin(__builtin_amdgcn_s_wait_asynccnt)
#define HAVE_ASYNC_LDS 1
#endif
#endif
#ifndef HAVE_ASYNC_LDS
#define HAVE_ASYNC_LDS 0
#endif

__device__ __forceinline__ v8f wmma_f16(v16h a, v16h b, v8f c) {
  return __builtin_amdgcn_wmma_f32_16x16x32_f16(false, a, false, b,
                                                (short)0, c, false, false);
}

// A-frag 16x32 from row-major f16 (lda halves). Layout:
// lanes 0-15 (M=lane): K0..15 ; lanes 16-31: K16..31 (contiguous).
__device__ __forceinline__ v16h load_a_rm(const _Float16* __restrict__ A,
                                          int lda, int m0, int kbase, int lane) {
  int r = lane & 15, h = lane >> 4;
  const _Float16* p = A + (size_t)(m0 + r) * lda + kbase + h * 16;
  v8h lo = *(const v8h*)p;
  v8h hi = *(const v8h*)(p + 8);
  v16h a;
#pragma unroll
  for (int e = 0; e < 8; ++e) { a[e] = lo[e]; a[8 + e] = hi[e]; }
  return a;
}

// B-frag from pre-swizzled weights: one 32B contiguous load per lane.
__device__ __forceinline__ v16h load_b_sw(const _Float16* __restrict__ Bsw,
                                          int kt, int nt, int lane) {
  return *(const v16h*)(Bsw + (size_t)(((kt * 16 + nt) * 32 + lane) * 16));
}

// ---------------------------------------------------------------------------
// 1) weight convert + swizzle into B-frag layout:
//    Bsw[((kt*16+nt)*32+lane)*16+e] = (f16) W[nt*16+(lane&15)][kt*32+(lane>>4)*16+e]
__global__ __launch_bounds__(256) void wswz_kernel(
    const float* __restrict__ wq, const float* __restrict__ ws,
    _Float16* __restrict__ wqsw, _Float16* __restrict__ wssw) {
  int i = blockIdx.x * 256 + threadIdx.x;       // 65536 total
  int e = i & 15, lane = (i >> 4) & 31, nt = (i >> 9) & 15, kt = i >> 13;
  int n = nt * 16 + (lane & 15);
  int c = kt * 32 + (lane >> 4) * 16 + e;
  wqsw[i] = (_Float16)wq[n * CDIM + c];
  wssw[i] = (_Float16)ws[n * CDIM + c];
}

// 2) fg/bg sums, softmax masks, pooling normalizers (shared across scales)
__global__ __launch_bounds__(256) void sums_kernel(
    const float* __restrict__ delta, float* __restrict__ inv,
    float* __restrict__ bfac, float* __restrict__ maskv) {
  int k = blockIdx.x, s = threadIdx.x;
  int p1 = s >> 4, p2 = s & 15;
  const float* dk = delta + (size_t)k * 65536;
  float raw = 0.f;
#pragma unroll
  for (int l = 0; l < 16; ++l) {
    int y = p1 * 4 + (l >> 2), x = p2 * 4 + (l & 3);
    raw += dk[y * 1024 + x * 4];                // nearest resize: stride-4
  }
  float rawb = 16.f - raw;
  int jf = k * 512 + s, jb = jf + 256;
  float cf = fmaxf(raw, 1e-6f), cb = fmaxf(rawb, 1e-6f);
  inv[jf] = 1.f / cf;       inv[jb] = 1.f / cb;
  bfac[jf] = raw / cf;      bfac[jb] = rawb / cb;
  maskv[jf] = (cf < 1.f) ? -1e30f : 0.f;
  maskv[jb] = (cb < 1.f) ? -1e30f : 0.f;
}

// 3) pooling of raw support images: P[j][c] (f16), one thread per patch s
__global__ __launch_bounds__(256) void pool_kernel(
    const float* __restrict__ x0, const float* __restrict__ x1,
    const float* __restrict__ x2, const float* __restrict__ x3,
    const float* __restrict__ delta, _Float16* __restrict__ pool16) {
  int k = blockIdx.x, scale = blockIdx.y, s = threadIdx.x;
  const float* xs[4] = {x0, x1, x2, x3};
  int p1 = s >> 4, p2 = s & 15;
  const float* dk = delta + (size_t)k * 65536;
  float fgw[16]; int pidx[16];
#pragma unroll
  for (int l = 0; l < 16; ++l) {
    int y = p1 * 4 + (l >> 2), x = p2 * 4 + (l & 3);
    fgw[l] = dk[y * 1024 + x * 4];
    pidx[l] = y * 64 + x;
  }
  const float* Xs = xs[scale] + (size_t)(1 + k) * CDIM * HW;   // support k
  _Float16* Pf = pool16 + ((size_t)scale * NJ + k * 512 + s) * CDIM;
  _Float16* Pb = Pf + (size_t)256 * CDIM;
  for (int c = 0; c < CDIM; ++c) {
    const float* xc = Xs + (size_t)c * HW;
    float af = 0.f, ab = 0.f;
#pragma unroll
    for (int l = 0; l < 16; ++l) {
      float v = xc[pidx[l]];
      af += fgw[l] * v;
      ab += (1.f - fgw[l]) * v;
    }
    Pf[c] = (_Float16)af;
    Pb[c] = (_Float16)ab;
  }
}

// 4) query image transpose: X[c][p] f32 -> qx16[p][c] f16 (coalesced both ways)
__global__ __launch_bounds__(256) void qtrans_kernel(
    const float* __restrict__ x0, const float* __restrict__ x1,
    const float* __restrict__ x2, const float* __restrict__ x3,
    _Float16* __restrict__ qx16) {
  __shared__ float tile[32][33];
  const float* xs[4] = {x0, x1, x2, x3};
  int scale = blockIdx.z;
  int p0 = blockIdx.x * 32, c0 = blockIdx.y * 32;
  int tx = threadIdx.x, ty = threadIdx.y;     // 32 x 8
  const float* X = xs[scale];                 // query image 0, channel-major
#pragma unroll
  for (int i = 0; i < 32; i += 8)
    tile[ty + i][tx] = X[(size_t)(c0 + ty + i) * HW + p0 + tx];
  __syncthreads();
#pragma unroll
  for (int i = 0; i < 32; i += 8)
    qx16[((size_t)scale * HW + p0 + ty + i) * CDIM + c0 + tx] =
        (_Float16)tile[tx][ty + i];
}

// 5) q projection GEMM: qf16[p][n] = qx @ Wq^T + bq  (M=4096,N=256,K=256)
__global__ __launch_bounds__(32) void qgemm_kernel(
    const _Float16* __restrict__ qx16, const _Float16* __restrict__ wqsw,
    const float* __restrict__ bq, _Float16* __restrict__ qf16) {
  int mt = blockIdx.x, ng = blockIdx.y, scale = blockIdx.z;
  int lane = threadIdx.x;
  const _Float16* A = qx16 + (size_t)scale * HW * CDIM;
  int m0 = mt * 16;
  v8f acc[4] = {};
  for (int kt = 0; kt < 8; ++kt) {
    v16h a = load_a_rm(A, CDIM, m0, kt * 32, lane);
#pragma unroll
    for (int g = 0; g < 4; ++g)
      acc[g] = wmma_f16(a, load_b_sw(wqsw, kt, ng * 4 + g, lane), acc[g]);
  }
  int col = lane & 15, half = lane >> 4;
#pragma unroll
  for (int g = 0; g < 4; ++g) {
    int n = ng * 64 + g * 16 + col;
    float bv = bq[n];
#pragma unroll
    for (int v = 0; v < 8; ++v) {
      int p = m0 + v + half * 8;
      qf16[((size_t)scale * HW + p) * CDIM + n] = (_Float16)(acc[g][v] + bv);
    }
  }
}

// 6) pool GEMM -> context vectors: c_s16[(scale*8+h)][j][d]  (M=2560,N=256,K=256)
__global__ __launch_bounds__(32) void poolgemm_kernel(
    const _Float16* __restrict__ pool16, const _Float16* __restrict__ wssw,
    const float* __restrict__ bs, const float* __restrict__ inv,
    const float* __restrict__ bfac, _Float16* __restrict__ cs16) {
  int mt = blockIdx.x, ng = blockIdx.y, scale = blockIdx.z;
  int lane = threadIdx.x;
  const _Float16* A = pool16 + (size_t)scale * NJ * CDIM;
  int m0 = mt * 16;
  v8f acc[4] = {};
  for (int kt = 0; kt < 8; ++kt) {
    v16h a = load_a_rm(A, CDIM, m0, kt * 32, lane);
#pragma unroll
    for (int g = 0; g < 4; ++g)
      acc[g] = wmma_f16(a, load_b_sw(wssw, kt, ng * 4 + g, lane), acc[g]);
  }
  int col = lane & 15, half = lane >> 4;
#pragma unroll
  for (int g = 0; g < 4; ++g) {
    int n0 = ng * 64 + g * 16;
    int n = n0 + col;
    int head = n0 >> 5, d = (n0 & 31) + col;
    float bv = bs[n];
#pragma unroll
    for (int v = 0; v < 8; ++v) {
      int j = m0 + v + half * 8;
      float val = acc[g][v] * inv[j] + bv * bfac[j];
      cs16[(((size_t)scale * HEADS + head) * NJ + j) * DDIM + d] = (_Float16)val;
    }
  }
}

// ---------------------------------------------------------------------------
// 7) attention: 8 waves share one head; context chunks staged to LDS in
// frag-major order [jl][lane][16] (32B contiguous per-lane frag reads).
__device__ __forceinline__ void stage_chunk(const _Float16* __restrict__ CS,
                                            _Float16* sdst, int ch, int tid) {
  int jl = tid >> 6, lanep = (tid >> 1) & 31, eh = tid & 1;
  const _Float16* src = CS + (size_t)(ch * CHUNK + jl * 16 + (lanep & 15)) * DDIM
                           + (lanep >> 4) * 16 + eh * 8;
  _Float16* dst = sdst + tid * 8;             // == ((jl*32+lanep)*2+eh)*8
#if HAVE_ASYNC_LDS
  __builtin_amdgcn_global_load_async_to_lds_b128(
      (__attribute__((address_space(1))) v4i*)const_cast<_Float16*>(src),
      (__attribute__((address_space(3))) v4i*)dst, 0, 0);
#else
  *(uint4*)dst = *(const uint4*)src;
#endif
}

__global__ __launch_bounds__(256) void attn_kernel(
    const _Float16* __restrict__ qf16, const _Float16* __restrict__ cs16,
    const float* __restrict__ maskv, float* __restrict__ cm) {
  __shared__ alignas(32) _Float16 sB[2][CHUNK * DDIM];
  int tid = threadIdx.x, wave = tid >> 5, lane = tid & 31;
  int head = blockIdx.y, scale = blockIdx.z;
  int mt = blockIdx.x * 8 + wave;
  int col = lane & 15, half = lane >> 4;

  const _Float16* Q = qf16 + (size_t)scale * HW * CDIM;
  v16h a = load_a_rm(Q, CDIM, mt * 16, head * DDIM, lane);  // 16 pos x 32 dims
  const _Float16* CS = cs16 + ((size_t)scale * HEADS + head) * NJ * DDIM;

  float mrun[8], sden[8], fnum[8];
#pragma unroll
  for (int v = 0; v < 8; ++v) { mrun[v] = -1e30f; sden[v] = 0.f; fnum[v] = 0.f; }

  stage_chunk(CS, &sB[0][0], 0, tid);
#if HAVE_ASYNC_LDS
  __builtin_amdgcn_s_wait_asynccnt(0);
#endif
  __syncthreads();

  for (int ch = 0; ch < NCHUNK; ++ch) {
    int cur = ch & 1;
    if (ch + 1 < NCHUNK) stage_chunk(CS, &sB[cur ^ 1][0], ch + 1, tid);
#pragma unroll
    for (int jl = 0; jl < CHUNK / 16; ++jl) {
      v16h b = *(const v16h*)(&sB[cur][(jl * 32 + lane) * 16]);
      v8f c = {};
      c = wmma_f16(a, b, c);
      int j = ch * CHUNK + jl * 16 + col;
      float mk = maskv[j];
      float fsel = ((j & 511) < 256) ? 1.f : 0.f;   // v[j]: fg=1, bg=0
#pragma unroll
      for (int v = 0; v < 8; ++v) {
        float val = c[v] + mk;
        float nm = fmaxf(mrun[v], val);
        float sc = __expf(mrun[v] - nm);
        float e  = __expf(val - nm);
        sden[v] = sden[v] * sc + e;
        fnum[v] = fnum[v] * sc + fsel * e;
        mrun[v] = nm;
      }
    }
#if HAVE_ASYNC_LDS
    __builtin_amdgcn_s_wait_asynccnt(0);
#endif
    __syncthreads();
  }
  // merge (m, s, f) across the 16 lanes of each half-wave
#pragma unroll
  for (int off = 1; off < 16; off <<= 1) {
#pragma unroll
    for (int v = 0; v < 8; ++v) {
      float om = __shfl_xor(mrun[v], off, 32);
      float os = __shfl_xor(sden[v], off, 32);
      float of = __shfl_xor(fnum[v], off, 32);
      float nm = fmaxf(mrun[v], om);
      float s0 = __expf(mrun[v] - nm), s1 = __expf(om - nm);
      sden[v] = sden[v] * s0 + os * s1;
      fnum[v] = fnum[v] * s0 + of * s1;
      mrun[v] = nm;
    }
  }
  if (col == 0) {
#pragma unroll
    for (int v = 0; v < 8; ++v) {
      int p = mt * 16 + v + half * 8;
      atomicAdd(&cm[(size_t)scale * HW + p], 0.125f * fnum[v] / sden[v]);
    }
  }
}

// ---------------------------------------------------------------------------
// Conv / GroupNorm tail (tiny workload, plain VALU)
__global__ __launch_bounds__(256) void conv2d_kernel(
    const float* __restrict__ in, const float* __restrict__ w,
    const float* __restrict__ b, float* __restrict__ out, int Cin, int Kh, int pad) {
  int oc = blockIdx.y;
  int p = blockIdx.x * 256 + threadIdx.x;
  int y = p >> 6, x = p & 63;
  float acc = b[oc];
  for (int ic = 0; ic < Cin; ++ic) {
    const float* inc = in + (size_t)ic * HW;
    const float* wc = w + ((size_t)oc * Cin + ic) * Kh * Kh;
    for (int ky = 0; ky < Kh; ++ky) {
      int iy = y + ky - pad;
      if (iy < 0 || iy >= 64) continue;
      for (int kx = 0; kx < Kh; ++kx) {
        int ix = x + kx - pad;
        if (ix < 0 || ix >= 64) continue;
        acc += inc[iy * 64 + ix] * wc[ky * Kh + kx];
      }
    }
  }
  out[(size_t)oc * HW + p] = acc;
}

__global__ __launch_bounds__(256) void gnstats_kernel(
    const float* __restrict__ X, float* __restrict__ mv, int cpg) {
  __shared__ float s1[256], s2[256];
  int g = blockIdx.x, tid = threadIdx.x;
  size_t n = (size_t)cpg * HW;
  const float* base = X + (size_t)g * n;
  float a = 0.f, b = 0.f;
  for (size_t i = tid; i < n; i += 256) { float v = base[i]; a += v; b += v * v; }
  s1[tid] = a; s2[tid] = b;
  __syncthreads();
  for (int off = 128; off > 0; off >>= 1) {
    if (tid < off) { s1[tid] += s1[tid + off]; s2[tid] += s2[tid + off]; }
    __syncthreads();
  }
  if (tid == 0) {
    float mean = s1[0] / (float)n;
    mv[g * 2] = mean;
    mv[g * 2 + 1] = s2[0] / (float)n - mean * mean;   // biased var (jnp)
  }
}

__global__ __launch_bounds__(256) void gnapply_kernel(
    const float* __restrict__ X, float* __restrict__ Y,
    const float* __restrict__ gamma, const float* __restrict__ beta,
    const float* __restrict__ mv, int cpg) {
  int c = blockIdx.y;
  int p = blockIdx.x * 256 + threadIdx.x;
  int g = c / cpg;
  float mean = mv[g * 2], var = mv[g * 2 + 1];
  float v = (X[(size_t)c * HW + p] - mean) * rsqrtf(var + 1e-5f)
            * gamma[c] + beta[c];
  Y[(size_t)c * HW + p] = fmaxf(v, 0.f);
}

// ---------------------------------------------------------------------------
extern "C" void kernel_launch(void* const* d_in, const int* in_sizes, int n_in,
                              void* d_out, int out_size, void* d_ws, size_t ws_size,
                              hipStream_t stream) {
  (void)in_sizes; (void)n_in; (void)out_size; (void)ws_size;
  const float* x0 = (const float*)d_in[0];
  const float* x1 = (const float*)d_in[1];
  const float* x2 = (const float*)d_in[2];
  const float* x3 = (const float*)d_in[3];
  const float* delta = (const float*)d_in[4];
  const float* wq = (const float*)d_in[5];
  const float* bq = (const float*)d_in[6];
  const float* ws = (const float*)d_in[7];
  const float* bs = (const float*)d_in[8];
  const float* cw1 = (const float*)d_in[9],  *cb1 = (const float*)d_in[10];
  const float* g1  = (const float*)d_in[11], *be1 = (const float*)d_in[12];
  const float* cw2 = (const float*)d_in[13], *cb2 = (const float*)d_in[14];
  const float* g2  = (const float*)d_in[15], *be2 = (const float*)d_in[16];
  const float* cw3 = (const float*)d_in[17], *cb3 = (const float*)d_in[18];
  const float* g3  = (const float*)d_in[19], *be3 = (const float*)d_in[20];

  char* base = (char*)d_ws;
  size_t off = 0;
  auto alloc = [&](size_t bytes) -> char* {
    off = (off + 255) & ~(size_t)255;
    char* p = base + off;
    off += bytes;
    return p;
  };
  _Float16* wqsw   = (_Float16*)alloc((size_t)CDIM * CDIM * 2);
  _Float16* wssw   = (_Float16*)alloc((size_t)CDIM * CDIM * 2);
  float*    invn   = (float*)alloc(NJ * 4);
  float*    bfac   = (float*)alloc(NJ * 4);
  float*    maskv  = (float*)alloc(NJ * 4);
  _Float16* pool16 = (_Float16*)alloc((size_t)4 * NJ * CDIM * 2);
  _Float16* qx16   = (_Float16*)alloc((size_t)4 * HW * CDIM * 2);
  _Float16* qf16   = (_Float16*)alloc((size_t)4 * HW * CDIM * 2);
  _Float16* cs16   = (_Float16*)alloc((size_t)4 * HEADS * NJ * DDIM * 2);
  float*    cm     = (float*)alloc((size_t)4 * HW * 4);
  float*    h1     = (float*)alloc((size_t)16 * HW * 4);
  float*    h2     = (float*)alloc((size_t)64 * HW * 4);
  float*    h3     = (float*)alloc((size_t)128 * HW * 4);
  float*    mv1    = (float*)alloc(8 * 4);
  float*    mv2    = (float*)alloc(8 * 4);
  float*    mv3    = (float*)alloc(8 * 4);

  (void)hipMemsetAsync(cm, 0, (size_t)4 * HW * 4, stream);

  wswz_kernel<<<dim3(256), dim3(256), 0, stream>>>(wq, ws, wqsw, wssw);
  sums_kernel<<<dim3(KIMG), dim3(SPAT), 0, stream>>>(delta, invn, bfac, maskv);
  pool_kernel<<<dim3(KIMG, 4), dim3(SPAT), 0, stream>>>(x0, x1, x2, x3, delta, pool16);
  qtrans_kernel<<<dim3(128, 8, 4), dim3(32, 8), 0, stream>>>(x0, x1, x2, x3, qx16);
  qgemm_kernel<<<dim3(HW / 16, 4, 4), dim3(32), 0, stream>>>(qx16, wqsw, bq, qf16);
  poolgemm_kernel<<<dim3(NJ / 16, 4, 4), dim3(32), 0, stream>>>(pool16, wssw, bs, invn, bfac, cs16);
  attn_kernel<<<dim3(HW / 16 / 8, HEADS, 4), dim3(256), 0, stream>>>(qf16, cs16, maskv, cm);

  conv2d_kernel<<<dim3(16, 16), dim3(256), 0, stream>>>(cm, cw1, cb1, h1, 4, 5, 2);
  gnstats_kernel<<<dim3(4), dim3(256), 0, stream>>>(h1, mv1, 4);
  gnapply_kernel<<<dim3(16, 16), dim3(256), 0, stream>>>(h1, h1, g1, be1, mv1, 4);

  conv2d_kernel<<<dim3(16, 64), dim3(256), 0, stream>>>(h1, cw2, cb2, h2, 16, 5, 2);
  gnstats_kernel<<<dim3(4), dim3(256), 0, stream>>>(h2, mv2, 16);
  gnapply_kernel<<<dim3(16, 64), dim3(256), 0, stream>>>(h2, h2, g2, be2, mv2, 16);

  conv2d_kernel<<<dim3(16, 128), dim3(256), 0, stream>>>(h2, cw3, cb3, h3, 64, 3, 1);
  gnstats_kernel<<<dim3(4), dim3(256), 0, stream>>>(h3, mv3, 32);
  gnapply_kernel<<<dim3(16, 128), dim3(256), 0, stream>>>(h3, (float*)d_out, g3, be3, mv3, 32);
}
